// MultilayerGRU_73443940761740
// MI455X (gfx1250) — compile-verified, matrix operand order checked by reading, not run on hardware
//
#include <hip/hip_runtime.h>
#include <math.h>

// ---- Problem sizes (fixed by the reference) ----
#define B_   64
#define S_   512
#define IN_  512
#define H_   1024
#define OUT_ 512

#define TPB       256          // 8 waves / block (wave32)
#define G_BLOCKS  96           // persistent grid: 4 M-tiles x 24 N-groups
#define LDS_PAD   8            // bf16 elements; makes row stride = 4 dwords (mod 64 banks)

#ifndef USE_ASYNC_LDS
#define USE_ASYNC_LDS 1        // gfx1250 GLOBAL_LOAD_ASYNC_TO_LDS path
#endif

typedef __attribute__((ext_vector_type(16))) __bf16 v16bf;
typedef __attribute__((ext_vector_type(8)))  float  v8f;

// ---------------------------------------------------------------------------
// WMMA fragment loads. Per CDNA5 ISA 7.12.2 (16-bit operands): lane L holds
// matrix row (L&15); K chunks [kb, kb+7] and [kb+16, kb+23], kb=(L>>4)*8.
// ---------------------------------------------------------------------------
__device__ __forceinline__ v16bf load_frag_g(const __bf16* __restrict__ base,
                                             int row0, int rowStride, int k0) {
    const int lane = threadIdx.x & 31;
    const int r    = lane & 15;
    const int kb   = (lane >> 4) << 3;
    const __bf16* p = base + (size_t)(row0 + r) * rowStride + k0 + kb;
    union { v16bf v; uint4 u[2]; } f;
    f.u[0] = *reinterpret_cast<const uint4*>(p);
    f.u[1] = *reinterpret_cast<const uint4*>(p + 16);
    return f.v;
}

__device__ __forceinline__ v16bf load_frag_lds(const __bf16* sm, int ldStride, int k0) {
    const int lane = threadIdx.x & 31;
    const int r    = lane & 15;
    const int kb   = (lane >> 4) << 3;
    const __bf16* p = sm + r * ldStride + k0 + kb;
    union { v16bf v; uint4 u[2]; } f;
    f.u[0] = *reinterpret_cast<const uint4*>(p);
    f.u[1] = *reinterpret_cast<const uint4*>(p + 16);
    return f.v;
}

__device__ __forceinline__ v8f wmma_bf16(v16bf a, v16bf b, v8f c) {
    return __builtin_amdgcn_wmma_f32_16x16x32_bf16(false, a, false, b,
                                                   (short)0, c, false, false);
}

__device__ __forceinline__ float sigmoidf_(float x) {
    return 1.0f / (1.0f + __expf(-x));
}

// ---------------------------------------------------------------------------
// Stage a 16-row x H tile (bf16) from global into LDS (padded row stride).
// Uses gfx1250 async global->LDS DMA when available; whole block cooperates.
// 2048 b128 transfers / 256 threads = 8 per thread, full EXEC throughout.
// ---------------------------------------------------------------------------
__device__ __forceinline__ void stage16xH(__bf16* sm, const __bf16* gbase) {
    const int per_row = H_ / 8;               // 128 x 16B chunks per row
#if USE_ASYNC_LDS
    const unsigned ldsBase = (unsigned)(size_t)sm;   // low 32b of generic ptr = LDS offset
#endif
#pragma unroll
    for (int i = 0; i < (16 * per_row) / TPB; ++i) {
        const int idx = i * TPB + threadIdx.x;
        const int r = idx / per_row;
        const int c = idx - r * per_row;
        const unsigned gOff = (unsigned)((r * H_ + c * 8) * 2);           // bytes
        const unsigned lOff = (unsigned)((r * (H_ + LDS_PAD) + c * 8) * 2);
#if USE_ASYNC_LDS
        asm volatile("global_load_async_to_lds_b128 %0, %1, %2"
                     :: "v"(ldsBase + lOff), "v"(gOff), "s"(gbase)
                     : "memory");
#else
        *reinterpret_cast<uint4*>(reinterpret_cast<char*>(sm) + lOff) =
            *reinterpret_cast<const uint4*>(reinterpret_cast<const char*>(gbase) + gOff);
#endif
    }
#if USE_ASYNC_LDS
    asm volatile("s_wait_asynccnt 0" ::: "memory");
#endif
    __syncthreads();
}

// ---------------------------------------------------------------------------
// Agent-scope grid barrier for the persistent scan kernel.
// ---------------------------------------------------------------------------
__device__ __forceinline__ void grid_barrier(unsigned* cnt, unsigned epoch) {
    __threadfence();                 // release: publish this thread's stores
    __syncthreads();
    if (threadIdx.x == 0) {
        __hip_atomic_fetch_add(cnt, 1u, __ATOMIC_RELAXED, __HIP_MEMORY_SCOPE_AGENT);
        const unsigned target = epoch * (unsigned)G_BLOCKS;
        while (__hip_atomic_load(cnt, __ATOMIC_RELAXED, __HIP_MEMORY_SCOPE_AGENT) < target) {
            __builtin_amdgcn_s_sleep(2);
        }
    }
    __syncthreads();
    __threadfence();                 // acquire: see other blocks' stores
}

// ---------------------------------------------------------------------------
// Phase 1: acc = x_t @ Wx^T + h @ Wh^T for this block's (m-tile, 8 n-tiles).
// A(h) comes from LDS; z -> sigmoid, r -> (sigmoid*h) bf16, g -> raw+bias.
// ---------------------------------------------------------------------------
__device__ __forceinline__ void phase1(const __bf16* xt, int xStride, int D,
                                       const __bf16* Wx, const __bf16* Wh,
                                       const float* bh, const float* hf,
                                       const __bf16* hbf,
                                       float* zbuf, __bf16* rhbf, float* xg,
                                       __bf16* sm, int m0, int ngrp) {
    stage16xH(sm, hbf + (size_t)m0 * H_);          // h rows for this m-tile

    const int ntile = ngrp * 8 + (threadIdx.x >> 5);   // 0..191
    const int n0 = ntile << 4;

    v8f acc = {};
    for (int k = 0; k < D; k += 32) {                  // input-side (global A)
        v16bf a = load_frag_g(xt, m0, xStride, k);
        v16bf b = load_frag_g(Wx, n0, D, k);
        acc = wmma_bf16(a, b, acc);
    }
    for (int k = 0; k < H_; k += 32) {                 // hidden-side (LDS A)
        v16bf a = load_frag_lds(sm, H_ + LDS_PAD, k);
        v16bf b = load_frag_g(Wh, n0, H_, k);
        acc = wmma_bf16(a, b, acc);
    }

    const int lane = threadIdx.x & 31;
    const int nn = lane & 15;
    const int mh = (lane >> 4) << 3;
    const int colAbs = n0 + nn;                        // gate is wave-uniform
    const float bias = bh[colAbs];
#pragma unroll
    for (int v = 0; v < 8; ++v) {
        const int row = m0 + v + mh;
        const float val = acc[v] + bias;
        if (colAbs < H_) {
            zbuf[row * H_ + colAbs] = sigmoidf_(val);
        } else if (colAbs < 2 * H_) {
            const int c = colAbs - H_;
            const float r = sigmoidf_(val);
            rhbf[row * H_ + c] = (__bf16)(r * hf[row * H_ + c]);
        } else {
            const int c = colAbs - 2 * H_;
            xg[row * H_ + c] = val;
        }
    }
}

// ---------------------------------------------------------------------------
// Phase 2: g = tanh(xg + (r*h) @ Whg^T); h = z*h + (1-z)*g.  Only n-groups 0..7
// (64 n-tiles) compute; all blocks still hit the grid barrier outside.
// ---------------------------------------------------------------------------
__device__ __forceinline__ void phase2(const __bf16* rhbf, const __bf16* Whg,
                                       const float* xg, const float* zbuf,
                                       float* hf, __bf16* hbf, __bf16* yt,
                                       __bf16* sm, int m0, int ngrp) {
    if (ngrp >= 8) return;                             // block-uniform
    stage16xH(sm, rhbf + (size_t)m0 * H_);

    const int ntile = ngrp * 8 + (threadIdx.x >> 5);   // 0..63
    const int n0 = ntile << 4;

    v8f acc = {};
    for (int k = 0; k < H_; k += 32) {
        v16bf a = load_frag_lds(sm, H_ + LDS_PAD, k);
        v16bf b = load_frag_g(Whg, n0, H_, k);
        acc = wmma_bf16(a, b, acc);
    }

    const int lane = threadIdx.x & 31;
    const int nn = lane & 15;
    const int mh = (lane >> 4) << 3;
#pragma unroll
    for (int v = 0; v < 8; ++v) {
        const int row = m0 + v + mh;
        const int col = n0 + nn;
        const int idx = row * H_ + col;
        const float g  = tanhf(xg[idx] + acc[v]);
        const float z  = zbuf[idx];
        const float hn = z * hf[idx] + (1.0f - z) * g;
        hf[idx]  = hn;
        hbf[idx] = (__bf16)hn;
        yt[(size_t)row * (S_ * H_) + col] = (__bf16)hn;   // yt = y + t*H
    }
}

// ---------------------------------------------------------------------------
// Persistent scan kernel: the entire 512-step, 2-layer recurrence in one
// launch; 4 grid barriers per timestep replace 4 host kernel launches.
// ---------------------------------------------------------------------------
__global__ void __launch_bounds__(TPB)
gru_scan(const __bf16* __restrict__ xb,
         const __bf16* __restrict__ Wx0, const __bf16* __restrict__ Wh0,
         const float*  __restrict__ bh0,
         const __bf16* __restrict__ Wx1, const __bf16* __restrict__ Wh1,
         const float*  __restrict__ bh1,
         float* h0f, __bf16* h0b, float* h1f, __bf16* h1b,
         __bf16* y0b, __bf16* y1b,
         float* zbuf, __bf16* rhb, float* xgb,
         unsigned* barCnt) {
    __shared__ __align__(16) __bf16 sm[16 * (H_ + LDS_PAD)];   // 33 KB

    const int m0   = (blockIdx.x & 3) << 4;   // this block's M-tile (B=64 -> 4)
    const int ngrp = blockIdx.x >> 2;         // 0..23 (8 n-tiles each)

    unsigned epoch = 0;
    for (int t = 0; t < S_; ++t) {
        phase1(xb + (size_t)t * IN_, S_ * IN_, IN_, Wx0, Wh0, bh0,
               h0f, h0b, zbuf, rhb, xgb, sm, m0, ngrp);
        grid_barrier(barCnt, ++epoch);
        phase2(rhb, Wh0 + (size_t)2 * H_ * H_, xgb, zbuf, h0f, h0b,
               y0b + (size_t)t * H_, sm, m0, ngrp);
        grid_barrier(barCnt, ++epoch);
        phase1(y0b + (size_t)t * H_, S_ * H_, H_, Wx1, Wh1, bh1,
               h1f, h1b, zbuf, rhb, xgb, sm, m0, ngrp);
        grid_barrier(barCnt, ++epoch);
        phase2(rhb, Wh1 + (size_t)2 * H_ * H_, xgb, zbuf, h1f, h1b,
               y1b + (size_t)t * H_, sm, m0, ngrp);
        grid_barrier(barCnt, ++epoch);
    }
}

// ---------------------------------------------------------------------------
// Output projection with 2x2 register blocking: 4 WMMA per 4 fragment loads.
// out[B*S, OUT] = y1 @ Why^T + by   (K = H)
// ---------------------------------------------------------------------------
__global__ void __launch_bounds__(TPB)
out_gemm(const __bf16* __restrict__ y1, const __bf16* __restrict__ Why,
         const float* __restrict__ by, float* __restrict__ out) {
    const int wave = (blockIdx.x << 3) + (threadIdx.x >> 5);
    const int nb = wave & 15;          // 16 groups of 2 N-tiles (OUT=512)
    const int mb = wave >> 4;          // 1024 groups of 2 M-tiles (B*S=32768)
    if (mb >= (B_ * S_) / 32) return;
    const int m0 = mb << 5;
    const int n0 = nb << 5;

    v8f a00 = {}, a01 = {}, a10 = {}, a11 = {};
    for (int k = 0; k < H_; k += 32) {
        v16bf fa0 = load_frag_g(y1, m0, H_, k);
        v16bf fa1 = load_frag_g(y1, m0 + 16, H_, k);
        v16bf fb0 = load_frag_g(Why, n0, H_, k);
        v16bf fb1 = load_frag_g(Why, n0 + 16, H_, k);
        a00 = wmma_bf16(fa0, fb0, a00);
        a01 = wmma_bf16(fa0, fb1, a01);
        a10 = wmma_bf16(fa1, fb0, a10);
        a11 = wmma_bf16(fa1, fb1, a11);
    }

    const int lane = threadIdx.x & 31;
    const int nn = lane & 15;
    const int mh = (lane >> 4) << 3;
    const float b0 = by[n0 + nn];
    const float b1 = by[n0 + 16 + nn];
#pragma unroll
    for (int v = 0; v < 8; ++v) {
        const size_t r0 = (size_t)(m0 + v + mh) * OUT_;
        const size_t r1 = (size_t)(m0 + 16 + v + mh) * OUT_;
        out[r0 + n0 + nn]      = a00[v] + b0;
        out[r0 + n0 + 16 + nn] = a01[v] + b1;
        out[r1 + n0 + nn]      = a10[v] + b0;
        out[r1 + n0 + 16 + nn] = a11[v] + b1;
    }
}

// ---------------------------------------------------------------------------
// Small helpers
// ---------------------------------------------------------------------------
__global__ void cast_f32_bf16(const float* __restrict__ s, __bf16* __restrict__ d, int n) {
    int i = blockIdx.x * blockDim.x + threadIdx.x;
    if (i < n) d[i] = (__bf16)s[i];
}

__global__ void init_h(const float* __restrict__ h0,
                       float* __restrict__ h0f, float* __restrict__ h1f,
                       __bf16* __restrict__ h0b, __bf16* __restrict__ h1b,
                       unsigned* __restrict__ barCnt) {
    int i = blockIdx.x * blockDim.x + threadIdx.x;
    if (i == 0) *barCnt = 0u;                       // reset grid-barrier counter
    if (i >= B_ * H_) return;
    int b = i / H_, j = i - b * H_;
    float a0 = h0[b * 2 * H_ + j];
    float a1 = h0[b * 2 * H_ + H_ + j];
    h0f[i] = a0; h1f[i] = a1;
    h0b[i] = (__bf16)a0; h1b[i] = (__bf16)a1;
}

__global__ void write_hidden(const float* __restrict__ h0f, const float* __restrict__ h1f,
                             float* __restrict__ outh) {
    int i = blockIdx.x * blockDim.x + threadIdx.x;
    if (i >= B_ * H_) return;
    int b = i / H_, j = i - b * H_;
    outh[b * 2 * H_ + j]      = h0f[i];
    outh[b * 2 * H_ + H_ + j] = h1f[i];
}

// ---------------------------------------------------------------------------
// Host-side orchestration
// ---------------------------------------------------------------------------
extern "C" void kernel_launch(void* const* d_in, const int* in_sizes, int n_in,
                              void* d_out, int out_size, void* d_ws, size_t ws_size,
                              hipStream_t stream) {
    (void)in_sizes; (void)n_in; (void)out_size; (void)ws_size;
    const float* x   = (const float*)d_in[0];
    const float* h0  = (const float*)d_in[1];
    const float* Wx0 = (const float*)d_in[2];
    const float* Wh0 = (const float*)d_in[3];
    const float* bh0 = (const float*)d_in[4];
    const float* Wx1 = (const float*)d_in[5];
    const float* Wh1 = (const float*)d_in[6];
    const float* bh1 = (const float*)d_in[7];
    const float* Why = (const float*)d_in[8];
    const float* by  = (const float*)d_in[9];
    float* out = (float*)d_out;

    char*  ws = (char*)d_ws;
    size_t o  = 0;
    auto alloc = [&](size_t bytes) -> char* {
        char* p = ws + o;
        o += (bytes + 255) & ~(size_t)255;
        return p;
    };
    __bf16* xb     = (__bf16*)alloc((size_t)B_ * S_ * IN_ * 2);
    __bf16* Wx0b   = (__bf16*)alloc((size_t)3 * H_ * IN_ * 2);
    __bf16* Wh0b   = (__bf16*)alloc((size_t)3 * H_ * H_ * 2);
    __bf16* Wx1b   = (__bf16*)alloc((size_t)3 * H_ * H_ * 2);
    __bf16* Wh1b   = (__bf16*)alloc((size_t)3 * H_ * H_ * 2);
    __bf16* Whyb   = (__bf16*)alloc((size_t)OUT_ * H_ * 2);
    __bf16* y0b    = (__bf16*)alloc((size_t)B_ * S_ * H_ * 2);
    __bf16* y1b    = (__bf16*)alloc((size_t)B_ * S_ * H_ * 2);
    float*  h0f    = (float*)alloc((size_t)B_ * H_ * 4);
    float*  h1f    = (float*)alloc((size_t)B_ * H_ * 4);
    __bf16* h0b    = (__bf16*)alloc((size_t)B_ * H_ * 2);
    __bf16* h1b    = (__bf16*)alloc((size_t)B_ * H_ * 2);
    float*  zbuf   = (float*)alloc((size_t)B_ * H_ * 4);
    __bf16* rhb    = (__bf16*)alloc((size_t)B_ * H_ * 2);
    float*  xgb    = (float*)alloc((size_t)B_ * H_ * 4);
    unsigned* barCnt = (unsigned*)alloc(256);

    auto blocks = [](size_t n) { return (unsigned)((n + 255) / 256); };

    // 1) bf16 conversion (weights stay hot in the 192MB L2 for the whole scan)
    cast_f32_bf16<<<blocks((size_t)B_ * S_ * IN_), TPB, 0, stream>>>(x, xb, B_ * S_ * IN_);
    cast_f32_bf16<<<blocks((size_t)3 * H_ * IN_), TPB, 0, stream>>>(Wx0, Wx0b, 3 * H_ * IN_);
    cast_f32_bf16<<<blocks((size_t)3 * H_ * H_), TPB, 0, stream>>>(Wh0, Wh0b, 3 * H_ * H_);
    cast_f32_bf16<<<blocks((size_t)3 * H_ * H_), TPB, 0, stream>>>(Wx1, Wx1b, 3 * H_ * H_);
    cast_f32_bf16<<<blocks((size_t)3 * H_ * H_), TPB, 0, stream>>>(Wh1, Wh1b, 3 * H_ * H_);
    cast_f32_bf16<<<blocks((size_t)OUT_ * H_), TPB, 0, stream>>>(Why, Whyb, OUT_ * H_);
    init_h<<<blocks((size_t)B_ * H_), TPB, 0, stream>>>(h0, h0f, h1f, h0b, h1b, barCnt);

    // 2) Entire 2-layer recurrence in ONE persistent kernel (grid barriers
    //    instead of 2048 launches).
    gru_scan<<<G_BLOCKS, TPB, 0, stream>>>(xb, Wx0b, Wh0b, bh0, Wx1b, Wh1b, bh1,
                                           h0f, h0b, h1f, h1b, y0b, y1b,
                                           zbuf, rhb, xgb, barCnt);

    // 3) Output projection (2x2-blocked WMMA GEMM) + hidden-state emit
    const unsigned OG_BLOCKS = ((B_ * S_ / 32) * (OUT_ / 32)) / 8;  // 2048
    out_gemm<<<OG_BLOCKS, TPB, 0, stream>>>(y1b, Whyb, by, out);
    write_hidden<<<blocks((size_t)B_ * H_), TPB, 0, stream>>>(
        h0f, h1f, out + (size_t)B_ * S_ * OUT_);
}
